// RegressionInstancesSharedCanonicalModule_19207093748139
// MI455X (gfx1250) — matrix operand order repaired and checked by reading, not compiled, not run on hardware
//
#include <hip/hip_runtime.h>
#include <hip/hip_bf16.h>

// ---------------------------------------------------------------------------
// Problem constants (from reference): B=2, I=63, H=W=320, DOWN=4 -> 80x80,
// HID=128, C=14, CLAMP_MIN=0.001
// ---------------------------------------------------------------------------
#define B_N    2
#define I_N    63
#define H_N    320
#define W_N    320
#define H4_N   80
#define W4_N   80
#define HW_N   (H4_N * W4_N)      // 6400
#define HID_N  128
#define NBIHW  (B_N * I_N * H_N * W_N)   // 12,902,400

typedef __attribute__((ext_vector_type(16))) _Float16 v16h;
typedef __attribute__((ext_vector_type(8)))  float    v8f;
typedef __attribute__((ext_vector_type(4)))  float    vf4;
typedef __attribute__((ext_vector_type(4)))  int      vi4;

// ---------------------------------------------------------------------------
// Kernel 1: pooled[b,i,c] = (sum_hw m_ds * F) / max(sum_hw m_ds, 1)
//
// Per workgroup: one (batch, 16-instance M-tile). 8 waves, each owning a
// K-segment of 800 (25 chunks of 32). Per wave: 8 f32 accumulators cover all
// 128 channels via v_wmma_f32_16x16x32_f16. Partials are reduced in LDS
// (deterministic tree: waves 4-7 add into waves 0-3, then a 4-way sum).
// B-tile raw data is software-pipelined (prefetch t+1 before WMMA t) so the
// b128 loads clause and overlap the matrix op instead of wait-stalling.
// ---------------------------------------------------------------------------
__global__ __launch_bounds__(256)
void pooled_wmma_kernel(const int* __restrict__ instances,
                        const float* __restrict__ fmap,
                        float* __restrict__ pooled)
{
    const int mtile = blockIdx.x;            // 0..3
    const int b     = blockIdx.y;            // 0..1
    const int wave  = threadIdx.x >> 5;      // 0..7  (K-segment owner)
    const int lane  = threadIdx.x & 31;      // wave32
    const int lo    = lane & 15;
    const int hi    = lane >> 4;

    __shared__ float red[4][16 * HID_N];     // 32 KB partial accumulators
    __shared__ float mpart[8][16];           // per-wave mask sums

    v8f zero = {};
    v8f acc[8];
#pragma unroll
    for (int t = 0; t < 8; ++t) acc[t] = zero;

    float msum = 0.0f;

    const int  inst      = mtile * 16 + lo;          // A-row this lane feeds
    const bool instValid = (inst < I_N);
    const long instBase  = instValid ? (long)(b * I_N + inst) * (H_N * W_N) : 0;

    // B base for this lane: channel (t*16 + lo), K offset 16*hi (contiguous 16)
    const float* fbase = fmap + (long)(b * HID_N + lo) * HW_N + 16 * hi;

    const int kseg = wave * 800;                     // this wave's K segment

    for (int c = 0; c < 25; ++c) {
        const int kc = kseg + c * 32;

        // ---- A: 16 raw strided mask ints (loads clause, one wait) ----
        int mraw[16];
#pragma unroll
        for (int run = 0; run < 2; ++run) {
            const int hw0 = kc + 16 * hi + run * 8;  // runs of 8 stay in a row
            const int h4  = hw0 / W4_N;
            const int w4  = hw0 % W4_N;
            const long gbase = instBase + (long)(h4 * 4) * W_N + (long)w4 * 4;
#pragma unroll
            for (int j = 0; j < 8; ++j)
                mraw[run * 8 + j] = instValid ? instances[gbase + j * 4] : 0;
        }
        v16h a;
#pragma unroll
        for (int e = 0; e < 16; ++e) {
            const float mv = (float)mraw[e];
            msum += mv;
            a[e] = (_Float16)mv;
        }

        // ---- B: 8 N-tiles, software-pipelined raw float4 loads ----
        vf4 cur[4], nxt[4];
        {
            const vf4* fp = (const vf4*)(fbase + kc);   // tile 0
#pragma unroll
            for (int q = 0; q < 4; ++q) cur[q] = fp[q];
        }
#pragma unroll
        for (int t = 0; t < 8; ++t) {
            if (t < 7) {
                const vf4* fp =
                    (const vf4*)(fbase + (long)(t + 1) * 16 * HW_N + kc);
#pragma unroll
                for (int q = 0; q < 4; ++q) nxt[q] = fp[q];
            }
            v16h bm;
#pragma unroll
            for (int q = 0; q < 4; ++q) {
#pragma unroll
                for (int e = 0; e < 4; ++e)
                    bm[4 * q + e] = (_Float16)cur[q][e];
            }
            acc[t] = __builtin_amdgcn_wmma_f32_16x16x32_f16(
                         false, a, false, bm, (short)0, acc[t], false, false);
#pragma unroll
            for (int q = 0; q < 4; ++q) cur[q] = nxt[q];
        }
    }

    // ---- per-wave mask sums: lane L and L+16 split K between them ----
    const float mtot = msum + __shfl_down(msum, 16, 32);     // lanes 0..15
    if (lane < 16) mpart[wave][lane] = mtot;

    // ---- cross-wave reduction in LDS (deterministic) ----
    if (wave < 4) {
#pragma unroll
        for (int t = 0; t < 8; ++t)
#pragma unroll
            for (int r = 0; r < 8; ++r)
                red[wave][(r + 8 * hi) * HID_N + t * 16 + lo] = acc[t][r];
    }
    __syncthreads();
    if (wave >= 4) {
#pragma unroll
        for (int t = 0; t < 8; ++t)
#pragma unroll
            for (int r = 0; r < 8; ++r)
                red[wave - 4][(r + 8 * hi) * HID_N + t * 16 + lo] += acc[t][r];
    }
    __syncthreads();

    // ---- final 4-way sum, denom, coalesced writeout ----
#pragma unroll
    for (int j = 0; j < 8; ++j) {
        const int p  = threadIdx.x + 256 * j;        // 0..2047
        const int il = p >> 7;                       // instance-in-tile
        const int inst2 = mtile * 16 + il;
        if (inst2 < I_N) {
            const float val = red[0][p] + red[1][p] + red[2][p] + red[3][p];
            float ms = 0.0f;
#pragma unroll
            for (int w = 0; w < 8; ++w) ms += mpart[w][il];
            pooled[(long)(b * I_N + inst2) * HID_N + (p & 127)] =
                val / fmaxf(ms, 1.0f);
        }
    }
}

// ---------------------------------------------------------------------------
// Kernel 2: scale/shift[b,i] = (pooled . W[label]) + b[label], masked by valid
// ---------------------------------------------------------------------------
__global__ __launch_bounds__(128)
void scale_shift_kernel(const float* __restrict__ pooled,
                        const int*   __restrict__ labels,
                        const float* __restrict__ Wsc, const float* __restrict__ bsc,
                        const float* __restrict__ Wsh, const float* __restrict__ bsh,
                        float* __restrict__ scaleOut, float* __restrict__ shiftOut)
{
    const int bi    = blockIdx.x;       // 0..125
    const int c     = threadIdx.x;      // channel
    const int label = labels[bi];

    const float p  = pooled[(long)bi * HID_N + c];
    float s1 = p * Wsc[label * HID_N + c];
    float s2 = p * Wsh[label * HID_N + c];

    __shared__ float r1[HID_N];
    __shared__ float r2[HID_N];
    r1[c] = s1; r2[c] = s2;
    __syncthreads();
#pragma unroll
    for (int off = 64; off > 0; off >>= 1) {
        if (c < off) { r1[c] += r1[c + off]; r2[c] += r2[c + off]; }
        __syncthreads();
    }
    if (c == 0) {
        const float valid = (label != 0) ? 1.0f : 0.0f;
        scaleOut[bi] = (r1[0] + bsc[label]) * valid;
        shiftOut[bi] = (r2[0] + bsh[label]) * valid;
    }
}

// ---------------------------------------------------------------------------
// Kernel 3: can_small[b,hw] = sum_c F[b,c,hw]*Wc[c] + bc   (2 x 6400)
// ---------------------------------------------------------------------------
__global__ __launch_bounds__(256)
void canonical_kernel(const float* __restrict__ fmap,
                      const float* __restrict__ Wc,
                      const float* __restrict__ bc,
                      float* __restrict__ canSmall)
{
    __shared__ float wl[HID_N];
    if (threadIdx.x < HID_N) wl[threadIdx.x] = Wc[threadIdx.x];
    __syncthreads();

    const int idx = blockIdx.x * 256 + threadIdx.x;   // 0..12799
    const int b   = idx / HW_N;
    const int hw  = idx % HW_N;

    float s = bc[0];
    const float* fp = fmap + (long)b * HID_N * HW_N + hw;
#pragma unroll 8
    for (int c = 0; c < HID_N; ++c) s += fp[(long)c * HW_N] * wl[c];
    canSmall[idx] = s;
}

// ---------------------------------------------------------------------------
// Kernel 4: memory-bound broadcast/upsample stage (~155 MB of traffic,
// ~7 us roofline at 23.3 TB/s). Streaming data uses non-temporal hints so
// write-once outputs / read-once masks don't churn L2.
//   canonical = can[b,h,w] * m * valid
//   depth     = max(canonical*scale + shift, 0.001)
// ---------------------------------------------------------------------------
__global__ __launch_bounds__(256)
void elementwise_kernel(const int*   __restrict__ instances,
                        const float* __restrict__ canSmall,
                        const int*   __restrict__ labels,
                        const float* __restrict__ scaleV,
                        const float* __restrict__ shiftV,
                        float* __restrict__ depthOut,
                        float* __restrict__ canOut)
{
    const int b  = blockIdx.z;
    const int i  = blockIdx.y;
    const int bi = b * I_N + i;

    const float s     = scaleV[bi];
    const float t     = shiftV[bi];
    const float valid = (labels[bi] != 0) ? 1.0f : 0.0f;

    const int q  = blockIdx.x * 256 + threadIdx.x;  // quad index 0..25599
    const int h  = q / (W_N / 4);                   // row 0..319
    const int qw = q % (W_N / 4);                   // quad-in-row 0..79

    const long base = ((long)bi * H_N + h) * W_N + (long)qw * 4;
    const vi4 mi = __builtin_nontemporal_load((const vi4*)(instances + base));

    const float can = canSmall[(long)b * HW_N + (h >> 2) * W4_N + qw] * valid;

    vf4 cf, dp;
#pragma unroll
    for (int e = 0; e < 4; ++e) {
        cf[e] = can * (float)mi[e];
        dp[e] = fmaxf(fmaf(cf[e], s, t), 0.001f);
    }

    __builtin_nontemporal_store(cf, (vf4*)(canOut   + base));
    __builtin_nontemporal_store(dp, (vf4*)(depthOut + base));
}

// ---------------------------------------------------------------------------
// Launch. Inputs (setup_inputs order):
//  0 depth  1 context  2 input_feature_map  3 bin_num  4 min_depth 5 max_depth
//  6 masks  7 instances  8 boxes  9 labels  10 W_scale 11 b_scale
//  12 W_shift 13 b_shift 14 Wc 15 bc
// Outputs (concat): depth_instances_r (NBIHW), canonical_full (NBIHW),
//                   scale (126), shift (126)
// ---------------------------------------------------------------------------
extern "C" void kernel_launch(void* const* d_in, const int* in_sizes, int n_in,
                              void* d_out, int out_size, void* d_ws, size_t ws_size,
                              hipStream_t stream)
{
    const float* fmap      = (const float*)d_in[2];
    const int*   instances = (const int*)  d_in[7];
    const int*   labels    = (const int*)  d_in[9];
    const float* Wsc       = (const float*)d_in[10];
    const float* bsc       = (const float*)d_in[11];
    const float* Wsh       = (const float*)d_in[12];
    const float* bsh       = (const float*)d_in[13];
    const float* Wc        = (const float*)d_in[14];
    const float* bc        = (const float*)d_in[15];

    float* depthOut = (float*)d_out;
    float* canOut   = depthOut + NBIHW;
    float* scaleOut = depthOut + 2L * NBIHW;
    float* shiftOut = scaleOut + (B_N * I_N);

    float* pooled   = (float*)d_ws;            // B*I*HID = 16128 floats
    float* canSmall = (float*)d_ws + 16384;    // B*HW    = 12800 floats

    // 1) masked-average pooling via WMMA (64 waves, K split 8-way per block)
    pooled_wmma_kernel<<<dim3(4, B_N), 256, 0, stream>>>(instances, fmap, pooled);

    // 2) per-instance scale/shift
    scale_shift_kernel<<<B_N * I_N, 128, 0, stream>>>(pooled, labels,
                                                      Wsc, bsc, Wsh, bsh,
                                                      scaleOut, shiftOut);

    // 3) canonical low-res map
    canonical_kernel<<<(B_N * HW_N) / 256, 256, 0, stream>>>(fmap, Wc, bc, canSmall);

    // 4) memory-bound broadcast/upsample stage (non-temporal streaming)
    elementwise_kernel<<<dim3((H_N * W_N / 4) / 256, I_N, B_N), 256, 0, stream>>>(
        instances, canSmall, labels, scaleOut, shiftOut, depthOut, canOut);
}
